// Dy_GrEncoder_7327214207524
// MI455X (gfx1250) — matrix operand order repaired and checked by reading, not compiled, not run on hardware
//
#include <hip/hip_runtime.h>

#define NN 100000
#define NE 3200000
#define H  128
#define NT 6250          // node tiles of 16 rows (NN/16)
#define NG 782           // tile groups of 8 tiles (ceil(NT/8))
#define NSTR 98          // group stride (8*98 = 784 blocks)

typedef __attribute__((ext_vector_type(16))) __bf16 v16bf;
typedef __attribute__((ext_vector_type(8)))  float  v8f;

union BfVec {
    v16bf v;
    unsigned u[8];
    __bf16 h[16];
};

__device__ __forceinline__ float bfbits2f(unsigned hw) {
    return __uint_as_float(hw << 16);
}
__device__ __forceinline__ unsigned f2bfbits(float f) {
    __bf16 b = (__bf16)f;
    return (unsigned)__builtin_bit_cast(unsigned short, b);
}
__device__ __forceinline__ float sigmoidf_(float x) {
    return 1.0f / (1.0f + __expf(-x));
}

__device__ __forceinline__ v8f wmma_bf16(v16bf a, v16bf b, v8f c) {
    return __builtin_amdgcn_wmma_f32_16x16x32_bf16(
        /*neg_a=*/false, a, /*neg_b=*/false, b,
        /*c_mod=*/(short)0, c, /*reuse_a=*/false, /*reuse_b=*/false);
}

// A-matrix (16x32 bf16) load from LDS row-major [16][128] tile.
// Lane l: row=l&15, g=l>>4. elems 0..15 -> K = kb*32+g*16+0..15
// (contiguous order, matching the B loads).
__device__ __forceinline__ v16bf load_a_tile(const __bf16* s, int lane, int kb) {
    BfVec a;
    const int row = lane & 15;
    const int g = lane >> 4;
    const unsigned* p = (const unsigned*)(s + row * H);
    const int b0 = (kb * 32 + g * 16) >> 1;
#pragma unroll
    for (int t = 0; t < 4; ++t) {
        a.u[t]     = p[b0 + t];
        a.u[4 + t] = p[b0 + 4 + t];
    }
    return a.v;
}

// Async global->LDS 16B copy, GVS form; tracked by ASYNCcnt.
__device__ __forceinline__ void async_b128(unsigned lds_byte, unsigned goff, const void* base) {
    asm volatile("global_load_async_to_lds_b128 %0, %1, %2"
                 :: "v"(lds_byte), "v"(goff), "s"(base) : "memory");
}
__device__ __forceinline__ void wait_async16() {
    asm volatile("s_wait_asynccnt 0x10" ::: "memory");
}
__device__ __forceinline__ void wait_async8() {
    asm volatile("s_wait_asynccnt 0x8" ::: "memory");
}
__device__ __forceinline__ void wait_async0() {
    asm volatile("s_wait_asynccnt 0x0" ::: "memory");
}

// ---------------------------------------------------------------------------
// K1: convert inputs to bf16 (W transposed so B loads are contiguous)
// ---------------------------------------------------------------------------
__global__ void convert_kernel(const float* __restrict__ x,
                               const float* __restrict__ W,
                               const float* __restrict__ gwih,
                               const float* __restrict__ gwhh,
                               const float* __restrict__ lwih,
                               __bf16* __restrict__ xb,
                               __bf16* __restrict__ Wtb,
                               __bf16* __restrict__ gwihb,
                               __bf16* __restrict__ gwhhb,
                               __bf16* __restrict__ lwihb) {
    const int nx = NN * H;
    const int nw = H * H;
    const int ng = 384 * H;
    const int nl = 512 * H;
    const int total = nx + nw + 2 * ng + nl;
    for (int i = blockIdx.x * blockDim.x + threadIdx.x; i < total;
         i += gridDim.x * blockDim.x) {
        if (i < nx) {
            xb[i] = (__bf16)x[i];
        } else if (i < nx + nw) {
            int t = i - nx;
            int j = t >> 7, k = t & 127;          // Wt[j][k] = W[k][j]
            Wtb[t] = (__bf16)W[k * H + j];
        } else if (i < nx + nw + ng) {
            int t = i - nx - nw;
            gwihb[t] = (__bf16)gwih[t];
        } else if (i < nx + nw + 2 * ng) {
            int t = i - nx - nw - ng;
            gwhhb[t] = (__bf16)gwhh[t];
        } else {
            int t = i - nx - nw - 2 * ng;
            lwihb[t] = (__bf16)lwih[t];
        }
    }
}

// ---------------------------------------------------------------------------
// K2: m = x @ W (bf16 WMMA, fp32 accum). Wtb (32KB) stays hot in WGP$.
// ---------------------------------------------------------------------------
__global__ __launch_bounds__(256) void gemm1_kernel(const __bf16* __restrict__ xb,
                                                    const __bf16* __restrict__ Wtb,
                                                    __bf16* __restrict__ mb) {
    __shared__ __bf16 sX[16 * H];
    const int tile = blockIdx.x;
    const int t = threadIdx.x;
    ((uint4*)sX)[t] = ((const uint4*)(xb + (size_t)tile * 16 * H))[t];
    __syncthreads();

    const int w = t >> 5, lane = t & 31;
    const int colL = lane & 15, g = lane >> 4;
    const int c = w * 16 + colL;

    v8f acc = {};
#pragma unroll
    for (int kb = 0; kb < 4; ++kb) {
        v16bf a = load_a_tile(sX, lane, kb);
        BfVec b;
        b.v = *(const v16bf*)(Wtb + (size_t)c * H + kb * 32 + g * 16);
        acc = wmma_bf16(a, b.v, acc);
    }
#pragma unroll
    for (int r = 0; r < 8; ++r) {
        int M = g * 8 + r;
        mb[((size_t)tile * 16 + M) * H + c] = (__bf16)acc[r];
    }
}

// ---------------------------------------------------------------------------
// K3/K4/K5: CSR build (degree count, prefix scan, fill)
// ---------------------------------------------------------------------------
__global__ void deg_kernel(const int* __restrict__ dst, int* __restrict__ deg) {
    int e = blockIdx.x * blockDim.x + threadIdx.x;
    if (e < NE) atomicAdd(&deg[dst[e]], 1);
}

__global__ void scan_kernel(const int* __restrict__ deg, int* __restrict__ offsets) {
    __shared__ int sdata[1024];
    __shared__ int srun;
    const int tid = threadIdx.x;
    if (tid == 0) { srun = 0; offsets[0] = 0; }
    __syncthreads();
    for (int base = 0; base < NN; base += 1024) {
        int i = base + tid;
        int v = (i < NN) ? deg[i] : 0;
        sdata[tid] = v;
        __syncthreads();
        for (int off = 1; off < 1024; off <<= 1) {
            int tmp = (tid >= off) ? sdata[tid - off] : 0;
            __syncthreads();
            sdata[tid] += tmp;
            __syncthreads();
        }
        if (i < NN) offsets[i + 1] = srun + sdata[tid];
        __syncthreads();
        if (tid == 0) srun += sdata[1023];
        __syncthreads();
    }
}

__global__ void fill_kernel(const int* __restrict__ src, const int* __restrict__ dst,
                            const int* __restrict__ offsets, int* __restrict__ cursor,
                            int* __restrict__ csr) {
    int e = blockIdx.x * blockDim.x + threadIdx.x;
    if (e < NE) {
        int d = dst[e];
        int pos = atomicAdd(&cursor[d], 1);
        csr[offsets[d] + pos] = src[e];
    }
}

// ---------------------------------------------------------------------------
// K6: pull-mode mean aggregation; one wave per dst node, L2-resident reads.
// ---------------------------------------------------------------------------
__global__ __launch_bounds__(256) void agg_kernel(const __bf16* __restrict__ mb,
                                                  const int* __restrict__ offsets,
                                                  const int* __restrict__ csr,
                                                  __bf16* __restrict__ aggb) {
    int v = blockIdx.x * 8 + (threadIdx.x >> 5);
    if (v >= NN) return;
    const int lane = threadIdx.x & 31;
    const int beg = offsets[v], end = offsets[v + 1];
    float a0 = 0.f, a1 = 0.f, a2 = 0.f, a3 = 0.f;
    for (int j = beg; j < end; ++j) {
        int s = csr[j];
        const unsigned* p = (const unsigned*)(mb + (size_t)s * H) + lane * 2;
        unsigned lo = p[0], hi = p[1];
        a0 += bfbits2f(lo & 0xffffu);
        a1 += bfbits2f(lo >> 16);
        a2 += bfbits2f(hi & 0xffffu);
        a3 += bfbits2f(hi >> 16);
    }
    float inv = 1.0f / fmaxf((float)(end - beg), 1.0f);
    a0 *= inv; a1 *= inv; a2 *= inv; a3 *= inv;
    unsigned* q = (unsigned*)(aggb + (size_t)v * H) + lane * 2;
    q[0] = f2bfbits(a0) | (f2bfbits(a1) << 16);
    q[1] = f2bfbits(a2) | (f2bfbits(a3) << 16);
}

// ---------------------------------------------------------------------------
// K7: GRU, column-sliced. Block owns cols [16c,16c+16); its 24KB weight slice
// lives in LDS; loops over groups of 8 node tiles with double-buffered
// async tile loads (global_load_async_to_lds_b128 + s_wait_asynccnt).
// LDS: tiles 2 x (8x4KB agg + 8x4KB x) = 128KB @ elem 0; weights @ elem 65536.
// ---------------------------------------------------------------------------
__global__ __launch_bounds__(256) void gru_kernel(
    const __bf16* __restrict__ aggb, const __bf16* __restrict__ xb,
    const __bf16* __restrict__ gwihb, const __bf16* __restrict__ gwhhb,
    const float* __restrict__ gbih, const float* __restrict__ gbhh,
    __bf16* __restrict__ hcb) {
    __shared__ __align__(16) __bf16 smem[77824];   // 65536 tiles + 12288 weights

    const int t = threadIdx.x;
    const int colBase = (blockIdx.x & 7) * 16;
    const int gstart = blockIdx.x >> 3;
    const unsigned ldsBase = (unsigned)(uintptr_t)smem;   // LDS aperture: low 32b

    // Prologue: async-fill buffer 0 with first tile group (32KB agg + 32KB x).
    {
        unsigned gof = (unsigned)gstart * 32768u + (unsigned)t * 128u;
        unsigned la = ldsBase + (unsigned)t * 128u;
#pragma unroll
        for (int j = 0; j < 8; ++j) {
            async_b128(la + j * 16u,          gof + j * 16u, aggb);
            async_b128(la + 32768u + j * 16u, gof + j * 16u, xb);
        }
    }

    // Stage weight slice: rows {colBase..+16} of r/z/n for W_ih and W_hh.
    // dwords [0..3071] = W_ih slice, [3072..6143] = W_hh slice.
    {
        const unsigned* gi = (const unsigned*)gwihb;
        const unsigned* gh = (const unsigned*)gwhhb;
        unsigned* sw = (unsigned*)(smem + 65536);
        for (int i = t; i < 6144; i += 256) {
            const unsigned* srcp = (i < 3072) ? gi : gh;
            int d = (i < 3072) ? i : i - 3072;
            int gate = d >> 10;
            int cl = (d >> 6) & 15;
            int kdw = d & 63;
            sw[i] = srcp[(gate * 128 + colBase + cl) * 64 + kdw];
        }
    }

    const int lane = t & 31, wv = t >> 5;
    const int colL = lane & 15, g16 = lane >> 4;
    const int c = colBase + colL;
    const float bir = gbih[c], biz = gbih[128 + c], bin = gbih[256 + c];
    const float bhr = gbhh[c], bhz = gbhh[128 + c], bhn = gbhh[256 + c];
    const __bf16* wih = smem + 65536;
    const __bf16* whh = smem + 65536 + 6144;

    int buf = 0;
    for (int g = gstart; g < NG; g += NSTR) {
        int gn = g + NSTR;
        if (gn < NG) {   // prefetch next group into the spent buffer
            unsigned gof = (unsigned)gn * 32768u + (unsigned)t * 128u;
            unsigned la = ldsBase + (unsigned)(buf ^ 1) * 65536u + (unsigned)t * 128u;
#pragma unroll
            for (int j = 0; j < 8; ++j) {
                async_b128(la + j * 16u,          gof + j * 16u, aggb);
                async_b128(la + 32768u + j * 16u, gof + j * 16u, xb);
            }
            wait_async16();   // current group's 16 ops (issued first) complete
        } else {
            wait_async0();
        }
        __syncthreads();

        const int tile = g * 8 + wv;
        if (tile < NT) {
            const __bf16* sA = smem + buf * 32768 + wv * 2048;
            const __bf16* sX = smem + buf * 32768 + 16384 + wv * 2048;
            v8f cir = {}, ciz = {}, cin = {}, chr_ = {}, chz = {}, chn = {};
#pragma unroll
            for (int kb = 0; kb < 4; ++kb) {
                v16bf aA = load_a_tile(sA, lane, kb);
                v16bf aX = load_a_tile(sX, lane, kb);
                const int kbase = kb * 32 + g16 * 16;
                BfVec b;
                b.v = *(const v16bf*)(wih + (0 * 16 + colL) * 128 + kbase);
                cir = wmma_bf16(aA, b.v, cir);
                b.v = *(const v16bf*)(wih + (1 * 16 + colL) * 128 + kbase);
                ciz = wmma_bf16(aA, b.v, ciz);
                b.v = *(const v16bf*)(wih + (2 * 16 + colL) * 128 + kbase);
                cin = wmma_bf16(aA, b.v, cin);
                b.v = *(const v16bf*)(whh + (0 * 16 + colL) * 128 + kbase);
                chr_ = wmma_bf16(aX, b.v, chr_);
                b.v = *(const v16bf*)(whh + (1 * 16 + colL) * 128 + kbase);
                chz = wmma_bf16(aX, b.v, chz);
                b.v = *(const v16bf*)(whh + (2 * 16 + colL) * 128 + kbase);
                chn = wmma_bf16(aX, b.v, chn);
            }
#pragma unroll
            for (int r = 0; r < 8; ++r) {
                int M = g16 * 8 + r;
                float rr = sigmoidf_(cir[r] + bir + chr_[r] + bhr);
                float zz = sigmoidf_(ciz[r] + biz + chz[r] + bhz);
                float nn = tanhf(cin[r] + bin + rr * (chn[r] + bhn));
                float xv = (float)sX[M * H + c];
                float hc = (1.0f - zz) * nn + zz * xv;
                hcb[((size_t)tile * 16 + M) * H + c] = (__bf16)hc;
            }
        }
        __syncthreads();   // all reads of `buf` done before next-iter prefetch
        buf ^= 1;
    }
}

// ---------------------------------------------------------------------------
// K8: LSTM (h0=c0=0: i/g/o gates only) + ReLU, column-sliced like K7.
// LDS: tiles 2 x 8x4KB = 64KB @ elem 0; 12KB weight slice @ elem 32768.
// ---------------------------------------------------------------------------
__global__ __launch_bounds__(256) void lstm_kernel(
    const __bf16* __restrict__ hcb, const __bf16* __restrict__ lwihb,
    const float* __restrict__ lbih, const float* __restrict__ lbhh,
    float* __restrict__ out) {
    __shared__ __align__(16) __bf16 smem[38912];   // 32768 tiles + 6144 weights

    const int t = threadIdx.x;
    const int colBase = (blockIdx.x & 7) * 16;
    const int gstart = blockIdx.x >> 3;
    const unsigned ldsBase = (unsigned)(uintptr_t)smem;

    {   // prologue async fill (32KB: 8 h_conv tiles)
        unsigned gof = (unsigned)gstart * 32768u + (unsigned)t * 128u;
        unsigned la = ldsBase + (unsigned)t * 128u;
#pragma unroll
        for (int j = 0; j < 8; ++j) async_b128(la + j * 16u, gof + j * 16u, hcb);
    }

    {   // stage weight slice: gates i(0), g(2), o(3)
        const unsigned* lw = (const unsigned*)lwihb;
        unsigned* sw = (unsigned*)(smem + 32768);
        for (int i = t; i < 3072; i += 256) {
            int gt = i >> 10, cl = (i >> 6) & 15, kdw = i & 63;
            int gsel = (gt == 0) ? 0 : gt + 1;     // 0, 2, 3
            sw[i] = lw[(gsel * 128 + colBase + cl) * 64 + kdw];
        }
    }

    const int lane = t & 31, wv = t >> 5;
    const int colL = lane & 15, g16 = lane >> 4;
    const int c = colBase + colL;
    const float bi = lbih[c] + lbhh[c];
    const float bg = lbih[256 + c] + lbhh[256 + c];
    const float bo = lbih[384 + c] + lbhh[384 + c];
    const __bf16* wl = smem + 32768;

    int buf = 0;
    for (int g = gstart; g < NG; g += NSTR) {
        int gn = g + NSTR;
        if (gn < NG) {
            unsigned gof = (unsigned)gn * 32768u + (unsigned)t * 128u;
            unsigned la = ldsBase + (unsigned)(buf ^ 1) * 32768u + (unsigned)t * 128u;
#pragma unroll
            for (int j = 0; j < 8; ++j) async_b128(la + j * 16u, gof + j * 16u, hcb);
            wait_async8();
        } else {
            wait_async0();
        }
        __syncthreads();

        const int tile = g * 8 + wv;
        if (tile < NT) {
            const __bf16* sA = smem + buf * 16384 + wv * 2048;
            v8f ci = {}, cg = {}, co = {};
#pragma unroll
            for (int kb = 0; kb < 4; ++kb) {
                v16bf aH = load_a_tile(sA, lane, kb);
                const int kbase = kb * 32 + g16 * 16;
                BfVec b;
                b.v = *(const v16bf*)(wl + (0 * 16 + colL) * 128 + kbase);
                ci = wmma_bf16(aH, b.v, ci);
                b.v = *(const v16bf*)(wl + (1 * 16 + colL) * 128 + kbase);
                cg = wmma_bf16(aH, b.v, cg);
                b.v = *(const v16bf*)(wl + (2 * 16 + colL) * 128 + kbase);
                co = wmma_bf16(aH, b.v, co);
            }
#pragma unroll
            for (int r = 0; r < 8; ++r) {
                int M = g16 * 8 + r;
                float cc = sigmoidf_(ci[r] + bi) * tanhf(cg[r] + bg);
                float h = sigmoidf_(co[r] + bo) * tanhf(cc);
                out[((size_t)tile * 16 + M) * H + c] = fmaxf(h, 0.0f);
            }
        }
        __syncthreads();
        buf ^= 1;
    }
}

// ---------------------------------------------------------------------------
extern "C" void kernel_launch(void* const* d_in, const int* in_sizes, int n_in,
                              void* d_out, int out_size, void* d_ws, size_t ws_size,
                              hipStream_t stream) {
    (void)in_sizes; (void)n_in; (void)out_size; (void)ws_size;
    const float* x    = (const float*)d_in[0];
    const int*   ei   = (const int*)d_in[1];      // [2, E] int32
    const float* W    = (const float*)d_in[2];
    const float* gwih = (const float*)d_in[3];
    const float* gwhh = (const float*)d_in[4];
    const float* gbih = (const float*)d_in[5];
    const float* gbhh = (const float*)d_in[6];
    const float* lwih = (const float*)d_in[7];
    // d_in[8] = lstm_w_hh: unused (h0 == 0)
    const float* lbih = (const float*)d_in[9];
    const float* lbhh = (const float*)d_in[10];
    float* out = (float*)d_out;

    const int* src = ei;
    const int* dst = ei + NE;

    char* ws = (char*)d_ws;
    size_t off = 0;
    auto carve = [&](size_t bytes) -> void* {
        off = (off + 255) & ~(size_t)255;
        void* p = ws + off;
        off += bytes;
        return p;
    };
    __bf16* xb      = (__bf16*)carve((size_t)NN * H * 2);
    __bf16* mb      = (__bf16*)carve((size_t)NN * H * 2);
    __bf16* aggb    = (__bf16*)carve((size_t)NN * H * 2);
    __bf16* hcb     = (__bf16*)carve((size_t)NN * H * 2);
    __bf16* Wtb     = (__bf16*)carve((size_t)H * H * 2);
    __bf16* gwihb   = (__bf16*)carve((size_t)384 * H * 2);
    __bf16* gwhhb   = (__bf16*)carve((size_t)384 * H * 2);
    __bf16* lwihb   = (__bf16*)carve((size_t)512 * H * 2);
    int*    deg     = (int*)carve((size_t)NN * 4);
    int*    cursor  = (int*)carve((size_t)NN * 4);
    int*    offsets = (int*)carve((size_t)(NN + 1) * 4);
    int*    csr     = (int*)carve((size_t)NE * 4);
    (void)carve(65536);   // slack: async tile prefetch of the ragged last group

    hipMemsetAsync(deg, 0, (size_t)NN * 4, stream);
    hipMemsetAsync(cursor, 0, (size_t)NN * 4, stream);

    convert_kernel<<<8192, 256, 0, stream>>>(x, W, gwih, gwhh, lwih,
                                             xb, Wtb, gwihb, gwhhb, lwihb);
    gemm1_kernel<<<NT, 256, 0, stream>>>(xb, Wtb, mb);
    deg_kernel<<<(NE + 255) / 256, 256, 0, stream>>>(dst, deg);
    scan_kernel<<<1, 1024, 0, stream>>>(deg, offsets);
    fill_kernel<<<(NE + 255) / 256, 256, 0, stream>>>(src, dst, offsets, cursor, csr);
    agg_kernel<<<(NN + 7) / 8, 256, 0, stream>>>(mb, offsets, csr, aggb);
    gru_kernel<<<8 * NSTR, 256, 0, stream>>>(aggb, xb, gwihb, gwhhb, gbih, gbhh, hcb);
    lstm_kernel<<<8 * NSTR, 256, 0, stream>>>(hcb, lwihb, lbih, lbhh, out);
}
